// GatedAttention_46583215293082
// MI455X (gfx1250) — matrix operand order repaired and checked by reading, not compiled, not run on hardware
//
#include <hip/hip_runtime.h>

#define SEQ    2048
#define DMODEL 2048
#define NHEADS 16
#define HDIM   128

typedef __attribute__((ext_vector_type(16))) __bf16 bf16x16;
typedef __attribute__((ext_vector_type(8)))  float  floatx8;
typedef unsigned short u16;

// ---------- bf16 helpers (bit-exact RNE truncation, no __bf16 arithmetic) ----------
__device__ __forceinline__ u16 f2bf(float f) {
    unsigned u = __float_as_uint(f);
    unsigned r = (u + 0x7FFFu + ((u >> 16) & 1u)) >> 16;
    return (u16)r;
}
__device__ __forceinline__ __bf16 bits2bf(u16 s) {
    __bf16 b; __builtin_memcpy(&b, &s, 2); return b;
}

// ---------- WMMA fragment loaders (CDNA5 ISA 7.12.2 layouts, wave32) ----------
// A fragment 16x32 (MxK), row-major source, leading dim ld.
// lane<16: M=lane,    K = {0..7, 16..23}
// lane>=16: M=lane-16, K = {8..15, 24..31}
__device__ __forceinline__ bf16x16 load_a_frag(const u16* base, int ld, int m0, int k0, int lane) {
    int m = lane & 15, kb = (lane >> 4) << 3;
    const u16* p = base + (size_t)(m0 + m) * ld + k0 + kb;
    bf16x16 r;
    __builtin_memcpy(&r, p, 16);                      // K = kb .. kb+7
    __builtin_memcpy(((char*)&r) + 16, p + 16, 16);   // K = kb+16 .. kb+23
    return r;
}
__device__ __forceinline__ bf16x16 load_a_frag(const float* base, int ld, int m0, int k0, int lane) {
    int m = lane & 15, kb = (lane >> 4) << 3;
    const float* p = base + (size_t)(m0 + m) * ld + k0 + kb;
    float t0[8], t1[8];
    __builtin_memcpy(t0, p, 32);
    __builtin_memcpy(t1, p + 16, 32);
    bf16x16 r;
#pragma unroll
    for (int i = 0; i < 8; ++i) { r[i] = bits2bf(f2bf(t0[i])); r[8 + i] = bits2bf(f2bf(t1[i])); }
    return r;
}
// B fragment 32x16 (KxN) where B[k][n] = Rows[n0+n][k0+k] (B = Rows^T), Rows row-major.
// lane<16: N=lane, K={0..7,16..23} ; lane>=16: N=lane-16, K={8..15,24..31} (same order as A).
__device__ __forceinline__ bf16x16 load_b_frag(const u16* base, int ld, int n0, int k0, int lane) {
    int n = lane & 15, kb = (lane >> 4) << 3;
    const u16* p = base + (size_t)(n0 + n) * ld + k0 + kb;
    bf16x16 r; __builtin_memcpy(&r, p, 16); __builtin_memcpy(((char*)&r) + 16, p + 16, 16); return r;
}
__device__ __forceinline__ bf16x16 load_b_frag(const float* base, int ld, int n0, int k0, int lane) {
    int n = lane & 15, kb = (lane >> 4) << 3;
    const float* p = base + (size_t)(n0 + n) * ld + k0 + kb;
    float t[16]; __builtin_memcpy(t, p, 32); __builtin_memcpy(t + 8, p + 16, 32);
    bf16x16 r;
#pragma unroll
    for (int i = 0; i < 16; ++i) r[i] = bits2bf(f2bf(t[i]));
    return r;
}
__device__ __forceinline__ floatx8 wmma_bf16(bf16x16 a, bf16x16 b, floatx8 c) {
    return __builtin_amdgcn_wmma_f32_16x16x32_bf16(false, a, false, b, (short)0, c, false, false);
}

// ================= Kernel 0: bulk fp32 -> bf16 conversion (memory bound, runs once) =========
__global__ __launch_bounds__(256) void cvt_bf16(const float* __restrict__ s,
                                                u16* __restrict__ d, int n) {
    int i = (blockIdx.x * 256 + threadIdx.x) * 8;
    if (i + 8 <= n) {
        float t[8];
        __builtin_memcpy(t, s + i, 32);
        u16 o[8];
#pragma unroll
        for (int j = 0; j < 8; ++j) o[j] = f2bf(t[j]);
        __builtin_memcpy(d + i, o, 16);
    }
}

// ================= Kernel 1: QKV projections  Y = x @ W^T =============
// z=0 -> Q [H][S][HDIM], z=1 -> K [H][S][HDIM], z=2 -> V^T [H][HDIM][S]
template <typename TX, typename TW>
__global__ __launch_bounds__(128) void qkv_gemm(const TX* __restrict__ x,
        const TW* __restrict__ qw, const TW* __restrict__ kw, const TW* __restrict__ vw,
        u16* __restrict__ qb, u16* __restrict__ kbuf, u16* __restrict__ vt) {
    const int lane = threadIdx.x & 31;
    const int wave = threadIdx.x >> 5;
    const int m0 = blockIdx.y * 64 + wave * 16;
    const int n0 = blockIdx.x * 64;
    const int z = blockIdx.z;
    const TW* w = (z == 0) ? qw : (z == 1) ? kw : vw;

    floatx8 acc[4] = {};
    for (int k0 = 0; k0 < DMODEL; k0 += 32) {
        bf16x16 a = load_a_frag(x, DMODEL, m0, k0, lane);
        if (k0 + 32 < DMODEL) {   // WGP-scope prefetch of next k-step (global_prefetch_b8)
            __builtin_prefetch(x + (size_t)(m0 + (lane & 15)) * DMODEL + k0 + 32, 0, 3);
            __builtin_prefetch(w + (size_t)(n0 + (lane & 15)) * DMODEL + k0 + 32, 0, 3);
        }
#pragma unroll
        for (int t = 0; t < 4; ++t) {
            bf16x16 b = load_b_frag(w, DMODEL, n0 + t * 16, k0, lane);
            acc[t] = wmma_bf16(a, b, acc[t]);
        }
    }
    const int n = lane & 15, mb = (lane >> 4) << 3;
#pragma unroll
    for (int t = 0; t < 4; ++t) {
        int e = n0 + t * 16 + n;
        int h = e >> 7, hd = e & 127;
#pragma unroll
        for (int j = 0; j < 8; ++j) {
            int s = m0 + mb + j;
            u16 v = f2bf(acc[t][j]);
            if (z == 2)      vt[((size_t)h * HDIM + hd) * SEQ + s] = v;
            else if (z == 0) qb[((size_t)h * SEQ + s) * HDIM + hd] = v;
            else             kbuf[((size_t)h * SEQ + s) * HDIM + hd] = v;
        }
    }
}

// ================= Kernel 2: causal flash attention + sigmoid gate (all-WMMA) ==============
template <typename TG>
__global__ __launch_bounds__(128) void flash_gate(
        const u16* __restrict__ qb, const u16* __restrict__ kbuf, const u16* __restrict__ vt,
        const TG* __restrict__ gate_w, const float* __restrict__ gate_bias,
        u16* __restrict__ att) {
    __shared__ u16 pst[4][16][32];   // per-wave P staging tile (f32->bf16 relayout via LDS)
    const int lane = threadIdx.x & 31;
    const int wave = threadIdx.x >> 5;
    const int h  = blockIdx.y;
    const int q0 = blockIdx.x * 64 + wave * 16;     // this wave's 16 query rows
    const u16* Qh = qb   + (size_t)h * SEQ * HDIM;
    const u16* Kh = kbuf + (size_t)h * SEQ * HDIM;
    const u16* Vh = vt   + (size_t)h * HDIM * SEQ;  // V^T: [HDIM][SEQ]
    const int n = lane & 15, mb = (lane >> 4) << 3;

    bf16x16 qf[4];                                   // Q 16x128 held in registers
#pragma unroll
    for (int ks = 0; ks < 4; ++ks) qf[ks] = load_a_frag(Qh, HDIM, q0, ks * 32, lane);

    floatx8 acc[8] = {};                             // 16x128 output accumulator
    float mi[8], li[8];
#pragma unroll
    for (int j = 0; j < 8; ++j) { mi[j] = -1e30f; li[j] = 0.f; }
    const float scale = 0.08838834764831845f;        // 1/sqrt(128)

    for (int kb = 0; kb <= q0 + 15; kb += 32) {      // causal: key blocks up to max query row
        floatx8 p[2];
#pragma unroll
        for (int t = 0; t < 2; ++t) {
            floatx8 sc = {};
            int kn0 = kb + t * 16;
#pragma unroll
            for (int ks = 0; ks < 4; ++ks) {
                bf16x16 bk = load_b_frag(Kh, HDIM, kn0, ks * 32, lane);  // scores = Q K^T
                sc = wmma_bf16(qf[ks], bk, sc);
            }
            int key = kn0 + n;
#pragma unroll
            for (int j = 0; j < 8; ++j) {
                float v = sc[j] * scale;
                p[t][j] = (key <= q0 + mb + j) ? v : -1e30f;   // causal mask
            }
        }
        // online softmax: row reductions across the 16 lanes holding each row
#pragma unroll
        for (int j = 0; j < 8; ++j) {
            float rm = fmaxf(p[0][j], p[1][j]);
#pragma unroll
            for (int msk = 1; msk < 16; msk <<= 1) rm = fmaxf(rm, __shfl_xor(rm, msk, 32));
            float mnew  = fmaxf(mi[j], rm);
            float alpha = __expf(mi[j] - mnew);
            float p0 = __expf(p[0][j] - mnew);
            float p1 = __expf(p[1][j] - mnew);
            p[0][j] = p0; p[1][j] = p1;
            float rs = p0 + p1;
#pragma unroll
            for (int msk = 1; msk < 16; msk <<= 1) rs += __shfl_xor(rs, msk, 32);
            li[j] = li[j] * alpha + rs;
            mi[j] = mnew;
#pragma unroll
            for (int t2 = 0; t2 < 8; ++t2) acc[t2][j] *= alpha;
        }
        // relayout P (C-layout f32) -> A-fragment bf16 through per-wave LDS tile
#pragma unroll
        for (int j = 0; j < 8; ++j) {
            pst[wave][mb + j][n]      = f2bf(p[0][j]);
            pst[wave][mb + j][16 + n] = f2bf(p[1][j]);
        }
        bf16x16 aP = load_a_frag(&pst[wave][0][0], 32, 0, 0, lane);
#pragma unroll
        for (int t2 = 0; t2 < 8; ++t2) {             // acc += P @ V  (V^T rows contiguous in keys)
            bf16x16 bv = load_b_frag(Vh, SEQ, t2 * 16, kb, lane);
            acc[t2] = wmma_bf16(aP, bv, acc[t2]);
        }
    }

    // gate = sigmoid(q @ gate_w^T + b), computed with WMMA from the same Q fragments
    floatx8 g[8] = {};
#pragma unroll
    for (int t2 = 0; t2 < 8; ++t2)
#pragma unroll
        for (int ks = 0; ks < 4; ++ks) {
            bf16x16 bg = load_b_frag(gate_w, HDIM, t2 * 16, ks * 32, lane);
            g[t2] = wmma_bf16(qf[ks], bg, g[t2]);
        }

#pragma unroll
    for (int t2 = 0; t2 < 8; ++t2) {
        int hd = t2 * 16 + n;
        float gb = gate_bias[hd];
#pragma unroll
        for (int j = 0; j < 8; ++j) {
            float gate = 1.f / (1.f + __expf(-(g[t2][j] + gb)));
            float val  = (acc[t2][j] / li[j]) * gate;
            att[(size_t)(q0 + mb + j) * DMODEL + h * HDIM + hd] = f2bf(val);
        }
    }
}

// ================= Kernel 3: output projection  out = attn @ o_w^T (fp32 out) ==============
template <typename TW>
__global__ __launch_bounds__(128) void out_gemm(const u16* __restrict__ att,
        const TW* __restrict__ ow, float* __restrict__ out) {
    const int lane = threadIdx.x & 31;
    const int wave = threadIdx.x >> 5;
    const int m0 = blockIdx.y * 64 + wave * 16;
    const int n0 = blockIdx.x * 64;
    floatx8 acc[4] = {};
    for (int k0 = 0; k0 < DMODEL; k0 += 32) {
        bf16x16 a = load_a_frag(att, DMODEL, m0, k0, lane);
        if (k0 + 32 < DMODEL) {
            __builtin_prefetch(att + (size_t)(m0 + (lane & 15)) * DMODEL + k0 + 32, 0, 3);
            __builtin_prefetch(ow  + (size_t)(n0 + (lane & 15)) * DMODEL + k0 + 32, 0, 3);
        }
#pragma unroll
        for (int t = 0; t < 4; ++t) {
            bf16x16 b = load_b_frag(ow, DMODEL, n0 + t * 16, k0, lane);
            acc[t] = wmma_bf16(a, b, acc[t]);
        }
    }
    const int n = lane & 15, mb = (lane >> 4) << 3;
#pragma unroll
    for (int t = 0; t < 4; ++t)
#pragma unroll
        for (int j = 0; j < 8; ++j)
            out[(size_t)(m0 + mb + j) * DMODEL + n0 + t * 16 + n] = acc[t][j];
}

extern "C" void kernel_launch(void* const* d_in, const int* in_sizes, int n_in,
                              void* d_out, int out_size, void* d_ws, size_t ws_size,
                              hipStream_t stream) {
    const float* x  = (const float*)d_in[0];
    const float* qw = (const float*)d_in[1];
    const float* kw = (const float*)d_in[2];
    const float* vw = (const float*)d_in[3];
    const float* ow = (const float*)d_in[4];
    const float* gw = (const float*)d_in[5];
    const float* gbv = (const float*)d_in[6];
    float* out = (float*)d_out;

    const size_t MAT  = (size_t)SEQ * DMODEL;     // elements per 2048x2048 matrix
    const size_t GMAT = (size_t)HDIM * HDIM;      // gate matrix elements

    // Always-needed intermediates (32 MB): Q, K, V^T, attn-out in bf16.
    u16* qb   = (u16*)d_ws;
    u16* kbuf = qb + MAT;
    u16* vt   = kbuf + MAT;
    u16* att  = vt + MAT;

    // Optional pre-converted bf16 operands (another ~40 MB).
    u16* xb  = att + MAT;
    u16* qwb = xb + MAT;
    u16* kwb = qwb + MAT;
    u16* vwb = kwb + MAT;
    u16* owb = vwb + MAT;
    u16* gwb = owb + MAT;
    const size_t need_full = (size_t)(9 * MAT + GMAT) * sizeof(u16);

    dim3 blk(128);
    if (ws_size >= need_full) {
        // One-time O(N^2) conversion so the O(N^3) WMMA loops are pure bf16 loads + v_wmma.
        const int CB = 256 * 8;
        cvt_bf16<<<dim3((MAT + CB - 1) / CB), dim3(256), 0, stream>>>(x,  xb,  (int)MAT);
        cvt_bf16<<<dim3((MAT + CB - 1) / CB), dim3(256), 0, stream>>>(qw, qwb, (int)MAT);
        cvt_bf16<<<dim3((MAT + CB - 1) / CB), dim3(256), 0, stream>>>(kw, kwb, (int)MAT);
        cvt_bf16<<<dim3((MAT + CB - 1) / CB), dim3(256), 0, stream>>>(vw, vwb, (int)MAT);
        cvt_bf16<<<dim3((MAT + CB - 1) / CB), dim3(256), 0, stream>>>(ow, owb, (int)MAT);
        cvt_bf16<<<dim3((GMAT + CB - 1) / CB), dim3(256), 0, stream>>>(gw, gwb, (int)GMAT);

        qkv_gemm<u16, u16><<<dim3(DMODEL / 64, SEQ / 64, 3), blk, 0, stream>>>(
            xb, qwb, kwb, vwb, qb, kbuf, vt);
        flash_gate<u16><<<dim3(SEQ / 64, NHEADS), blk, 0, stream>>>(qb, kbuf, vt, gwb, gbv, att);
        out_gemm<u16><<<dim3(DMODEL / 64, SEQ / 64), blk, 0, stream>>>(att, owb, out);
    } else {
        // Fallback: convert fp32 -> bf16 on the fly inside the fragment loaders.
        qkv_gemm<float, float><<<dim3(DMODEL / 64, SEQ / 64, 3), blk, 0, stream>>>(
            x, qw, kw, vw, qb, kbuf, vt);
        flash_gate<float><<<dim3(SEQ / 64, NHEADS), blk, 0, stream>>>(qb, kbuf, vt, gw, gbv, att);
        out_gemm<float><<<dim3(DMODEL / 64, SEQ / 64), blk, 0, stream>>>(att, ow, out);
    }
}
